// NodeUpdateNetwork_67190468378756
// MI455X (gfx1250) — matrix operand order repaired and checked by reading, not compiled, not run on hardware
//
#include <hip/hip_runtime.h>
#include <hip/hip_bf16.h>
#include <stdint.h>

// ---------------------------------------------------------------------------
// Types / helpers for CDNA5 (gfx1250) bf16 WMMA: D(16x16 f32) = A(16x32) x B(32x16) + C
// ---------------------------------------------------------------------------
typedef __attribute__((ext_vector_type(16))) __bf16 v16bf;
typedef __attribute__((ext_vector_type(8)))  float  v8f;

union FragU  { uint4 u[2]; unsigned short s[16]; v16bf bf; };
union Half8U { uint4 u;    unsigned short s[8]; };

__device__ __forceinline__ unsigned short f2bf(float f) {
    unsigned u = __builtin_bit_cast(unsigned, f);
    u += 0x7FFFu + ((u >> 16) & 1u);            // round-to-nearest-even
    return (unsigned short)(u >> 16);
}
__device__ __forceinline__ float bf2f(unsigned short h) {
    unsigned u = ((unsigned)h) << 16;
    return __builtin_bit_cast(float, u);
}

__device__ __forceinline__ v8f wmma_bf16(v16bf a, v16bf b, v8f c) {
    return __builtin_amdgcn_wmma_f32_16x16x32_bf16(false, a, false, b, (short)0, c,
                                                   false, false);
}

// A-fragment (16x32, M x K), source row-major [M][K], ld in elements.
// ISA layout: lanes 0-15 M=lane, K-chunks {h*8..h*8+7, 16+h*8..}; h = lane>=16.
__device__ __forceinline__ v16bf load_a_frag(const unsigned short* base, int ld, int lane) {
    const int m = lane & 15, h = (lane >> 4) & 1;
    const unsigned short* p = base + (size_t)m * ld + h * 8;
    FragU f;
    f.u[0] = *(const uint4*)(p);
    f.u[1] = *(const uint4*)(p + 16);
    return f.bf;
}

// B-fragment (32x16, K x N), source: per output-column n, K contiguous (S[n][k], ld).
// ISA layout: N=lane&15, K-chunks {h*8..h*8+7, 16+h*8..}; h = lane>=16.
__device__ __forceinline__ v16bf load_b_frag(const unsigned short* base, int ld, int lane) {
    const int n = lane & 15, h = (lane >> 4) & 1;
    const unsigned short* p = base + (size_t)n * ld + h * 8;
    FragU f;
    f.u[0] = *(const uint4*)(p);
    f.u[1] = *(const uint4*)(p + 16);
    return f.bf;
}

// A-fragment with fused BatchNorm + LeakyReLU applied elementwise (channel = K index).
__device__ __forceinline__ v16bf load_a_bn(const unsigned short* base, int ld, int lane,
                                           const float* s0, const float* b0, int kbase) {
    const int m = lane & 15, h = (lane >> 4) & 1;
    const unsigned short* p = base + (size_t)m * ld + h * 8;
    FragU raw, r;
    raw.u[0] = *(const uint4*)(p);
    raw.u[1] = *(const uint4*)(p + 16);
#pragma unroll
    for (int j = 0; j < 8; ++j) {
        int c0 = kbase + h * 8 + j, c1 = c0 + 16;
        float z0 = fmaf(bf2f(raw.s[j]),     s0[c0], b0[c0]);
        float z1 = fmaf(bf2f(raw.s[8 + j]), s0[c1], b0[c1]);
        z0 = z0 > 0.f ? z0 : 0.01f * z0;
        z1 = z1 > 0.f ? z1 : 0.01f * z1;
        r.s[j]     = f2bf(z0);
        r.s[8 + j] = f2bf(z1);
    }
    return r.bf;
}

// ---------------------------------------------------------------------------
// K_w: weights fp32 -> bf16
// ---------------------------------------------------------------------------
__global__ __launch_bounds__(256) void k_weights(const float* __restrict__ w0,
                                                 const float* __restrict__ w1,
                                                 unsigned short* __restrict__ w0bf,
                                                 unsigned short* __restrict__ w1bf) {
    int t = blockIdx.x * 256 + threadIdx.x;
    int stride = gridDim.x * 256;
    for (int i = t; i < 256 * 96;  i += stride) w0bf[i] = f2bf(w0[i]);
    for (int i = t; i < 128 * 256; i += stride) w1bf[i] = f2bf(w1[i]);
}

// ---------------------------------------------------------------------------
// K0: node_feat[b][n][f] fp32 -> xT[b][n][0..127] bf16 (node part of concat)
//                             -> nodeT[b][f][n]   bf16 (A-operand of aggregation)
// ---------------------------------------------------------------------------
__global__ __launch_bounds__(256) void k0_prep(const float* __restrict__ node,
                                               unsigned short* __restrict__ xT,
                                               unsigned short* __restrict__ nodeT) {
    __shared__ unsigned short L[128 * 130];   // padded to dodge bank conflicts
    const int b = blockIdx.x, t = threadIdx.x;
    const float4* src = (const float4*)(node + (size_t)b * 128 * 128);
#pragma unroll 4
    for (int i = 0; i < 16; ++i) {
        int vi = i * 256 + t;                 // 4096 float4 total
        float4 v = src[vi];
        int base = vi * 4, n = base >> 7, f = base & 127;
        unsigned short s0 = f2bf(v.x), s1 = f2bf(v.y), s2 = f2bf(v.z), s3 = f2bf(v.w);
        L[n * 130 + f] = s0; L[n * 130 + f + 1] = s1;
        L[n * 130 + f + 2] = s2; L[n * 130 + f + 3] = s3;
        *(ushort4*)(xT + ((size_t)b * 128 + n) * 384 + f) = make_ushort4(s0, s1, s2, s3);
    }
    __syncthreads();
#pragma unroll 4
    for (int i = 0; i < 16; ++i) {
        int vi = i * 256 + t;
        int base = vi * 4, f = base >> 7, n0 = base & 127;
        ushort4 pk = make_ushort4(L[(n0 + 0) * 130 + f], L[(n0 + 1) * 130 + f],
                                  L[(n0 + 2) * 130 + f], L[(n0 + 3) * 130 + f]);
        *(ushort4*)(nodeT + ((size_t)b * 128 + f) * 128 + n0) = pk;
    }
}

// ---------------------------------------------------------------------------
// K1: per (b,k): mask diag, L1-normalize rows of edge; aggrT = nodeT @ e^T via WMMA
//     writes xT[b][r][128 + k*128 + f]  (aggr part of concat, bf16)
// ---------------------------------------------------------------------------
__global__ __launch_bounds__(256) void k1_aggr(const float* __restrict__ edge,
                                               const unsigned short* __restrict__ nodeT,
                                               unsigned short* __restrict__ xT) {
    __shared__ unsigned short E[128 * 128];   // normalized edges, bf16, [r][m]
    __shared__ float red[256];
    __shared__ float inv[128];
    const int bk = blockIdx.x, b = bk >> 1, k = bk & 1, t = threadIdx.x;
    const float* eg = edge + (size_t)bk * 16384;

    // Phase A: load fp32, mask diagonal, store bf16 to LDS, row abs-sum (2 thr/row)
    {
        const int row = t >> 1, hm = t & 1;
        float s = 0.f;
#pragma unroll 4
        for (int i = 0; i < 16; ++i) {
            const int m0 = hm * 64 + i * 4;
            float4 v = *(const float4*)(eg + row * 128 + m0);
            float a0 = (m0 + 0 == row) ? 0.f : v.x;
            float a1 = (m0 + 1 == row) ? 0.f : v.y;
            float a2 = (m0 + 2 == row) ? 0.f : v.z;
            float a3 = (m0 + 3 == row) ? 0.f : v.w;
            s += fabsf(a0) + fabsf(a1) + fabsf(a2) + fabsf(a3);
            *(ushort4*)(E + row * 128 + m0) = make_ushort4(f2bf(a0), f2bf(a1),
                                                           f2bf(a2), f2bf(a3));
        }
        red[t] = s;
    }
    __syncthreads();
    if (t < 128) inv[t] = 1.f / fmaxf(red[2 * t] + red[2 * t + 1], 1e-12f);
    __syncthreads();
    // Phase B: rescale rows in LDS
#pragma unroll 4
    for (int i = 0; i < 16; ++i) {
        int base = (i * 256 + t) * 4, r = base >> 7;
        float iv = inv[r];
        ushort4 v = *(ushort4*)(E + base);
        *(ushort4*)(E + base) = make_ushort4(f2bf(bf2f(v.x) * iv), f2bf(bf2f(v.y) * iv),
                                             f2bf(bf2f(v.z) * iv), f2bf(bf2f(v.w) * iv));
    }
    __syncthreads();

    // Phase C: D^T[f][r] = nodeT[f][m] @ e^T[m][r];  M=f, N=r, K=m (128 = 4 k-steps)
    const int wv = t >> 5, lane = t & 31, h = (lane >> 4) & 1, ln = lane & 15;
    const int ft = wv * 16;
    const unsigned short* Ab = nodeT + (size_t)b * 16384 + (size_t)ft * 128;
    v8f acc[8];
    for (int nt = 0; nt < 8; ++nt)
        for (int e = 0; e < 8; ++e) acc[nt][e] = 0.f;
#pragma unroll
    for (int ks = 0; ks < 4; ++ks) {
        v16bf a = load_a_frag(Ab + ks * 32, 128, lane);
#pragma unroll
        for (int nt = 0; nt < 8; ++nt) {
            v16bf bb = load_b_frag(E + nt * 16 * 128 + ks * 32, 128, lane);
            acc[nt] = wmma_bf16(a, bb, acc[nt]);
        }
    }
    // D element v: f = ft + 8h + v (contiguous!), r = ntile + ln -> one b128 store/tile
    const int col0 = 128 + k * 128 + ft + 8 * h;
#pragma unroll
    for (int nt = 0; nt < 8; ++nt) {
        Half8U pk;
#pragma unroll
        for (int v = 0; v < 8; ++v) pk.s[v] = f2bf(acc[nt][v]);
        *(uint4*)(xT + ((size_t)b * 128 + nt * 16 + ln) * 384 + col0) = pk.u;
    }
}

// ---------------------------------------------------------------------------
// K2: grouped conv0: y0T[b][n][cout] = (W0_g @ x_g)^T computed as x_g^T @ W0_g^T
//     M=n(128), N=cout(256, 16 tiles, group=ct/4), K=96 per group.
//     Deterministic per-block channel sum/sumsq partials.
// ---------------------------------------------------------------------------
__global__ __launch_bounds__(256) void k2_layer0(const unsigned short* __restrict__ xT,
                                                 const unsigned short* __restrict__ w0bf,
                                                 unsigned short* __restrict__ y0T,
                                                 float* __restrict__ partS0,
                                                 float* __restrict__ partQ0) {
    __shared__ float sred[16 * 256];
    __shared__ float qred[16 * 256];
    const int b = blockIdx.x, t = threadIdx.x;
    __builtin_prefetch(w0bf + t * 96, 0, 3);
    const int wv = t >> 5, lane = t & 31, h = (lane >> 4) & 1, ln = lane & 15;
    const unsigned short* Xb = xT + (size_t)b * 49152 + (size_t)wv * 16 * 384;

    for (int g = 0; g < 4; ++g) {
        v8f acc[4];
        for (int j = 0; j < 4; ++j)
            for (int e = 0; e < 8; ++e) acc[j][e] = 0.f;
#pragma unroll
        for (int ks = 0; ks < 3; ++ks) {
            v16bf a = load_a_frag(Xb + g * 96 + ks * 32, 384, lane);
#pragma unroll
            for (int c4 = 0; c4 < 4; ++c4) {
                int ct = g * 4 + c4;
                v16bf bb = load_b_frag(w0bf + (size_t)ct * 16 * 96 + ks * 32, 96, lane);
                acc[c4] = wmma_bf16(a, bb, acc[c4]);
            }
        }
#pragma unroll
        for (int c4 = 0; c4 < 4; ++c4) {
            const int c = (g * 4 + c4) * 16 + ln;
            float s = 0.f, q = 0.f;
#pragma unroll
            for (int v = 0; v < 8; ++v) { float x = acc[c4][v]; s += x; q += x * x; }
            sred[(wv * 2 + h) * 256 + c] = s;
            qred[(wv * 2 + h) * 256 + c] = q;
#pragma unroll
            for (int v = 0; v < 8; ++v)
                y0T[((size_t)b * 128 + wv * 16 + v + 8 * h) * 256 + c] = f2bf(acc[c4][v]);
        }
    }
    __syncthreads();
    {   // fixed-order cross-wave reduction -> deterministic partials
        float S = 0.f, Q = 0.f;
        for (int sl = 0; sl < 16; ++sl) { S += sred[sl * 256 + t]; Q += qred[sl * 256 + t]; }
        partS0[(size_t)b * 256 + t] = S;
        partQ0[(size_t)b * 256 + t] = Q;
    }
}

// ---------------------------------------------------------------------------
// K3/K5: finalize BN: scale = gamma*rsqrt(var+eps), bias = beta - mean*scale
// ---------------------------------------------------------------------------
__global__ void k_finalize(const float* __restrict__ partS, const float* __restrict__ partQ,
                           const float* __restrict__ gamma, const float* __restrict__ beta,
                           float* __restrict__ scale, float* __restrict__ bias,
                           int C, int nPart, float invN) {
    int c = blockIdx.x * blockDim.x + threadIdx.x;
    if (c >= C) return;
    float S = 0.f, Q = 0.f;
    for (int p = 0; p < nPart; ++p) { S += partS[(size_t)p * C + c]; Q += partQ[(size_t)p * C + c]; }
    float mean = S * invN;
    float var  = Q * invN - mean * mean;
    float sc = gamma[c] * rsqrtf(var + 1e-5f);
    scale[c] = sc;
    bias[c]  = beta[c] - mean * sc;
}

// ---------------------------------------------------------------------------
// K4: conv1 (groups=1): y1T = z^T @ W1^T, z = leaky(bn0(y0)) fused into A-frag load.
//     M=n(128), N=cout(128), K=256. Raw (pre-BN1) result -> d_out directly.
// ---------------------------------------------------------------------------
__global__ __launch_bounds__(256) void k4_layer1(const unsigned short* __restrict__ y0T,
                                                 const unsigned short* __restrict__ w1bf,
                                                 const float* __restrict__ scale0,
                                                 const float* __restrict__ bias0,
                                                 float* __restrict__ out,
                                                 float* __restrict__ partS1,
                                                 float* __restrict__ partQ1) {
    __shared__ float s0[256], b0[256];
    __shared__ float sred[16 * 128], qred[16 * 128];
    const int b = blockIdx.x, t = threadIdx.x;
    s0[t] = scale0[t];
    b0[t] = bias0[t];
    __builtin_prefetch(w1bf + t * 128, 0, 3);
    __syncthreads();

    const int wv = t >> 5, lane = t & 31, h = (lane >> 4) & 1, ln = lane & 15;
    const unsigned short* Yb = y0T + (size_t)b * 32768 + (size_t)wv * 16 * 256;
    v8f acc[8];
    for (int ct = 0; ct < 8; ++ct)
        for (int e = 0; e < 8; ++e) acc[ct][e] = 0.f;
#pragma unroll
    for (int ks = 0; ks < 8; ++ks) {
        v16bf a = load_a_bn(Yb + ks * 32, 256, lane, s0, b0, ks * 32);
#pragma unroll
        for (int ct = 0; ct < 8; ++ct) {
            v16bf bb = load_b_frag(w1bf + (size_t)ct * 16 * 256 + ks * 32, 256, lane);
            acc[ct] = wmma_bf16(a, bb, acc[ct]);
        }
    }
#pragma unroll
    for (int ct = 0; ct < 8; ++ct) {
        const int c = ct * 16 + ln;
        float s = 0.f, q = 0.f;
#pragma unroll
        for (int v = 0; v < 8; ++v) { float x = acc[ct][v]; s += x; q += x * x; }
        sred[(wv * 2 + h) * 128 + c] = s;
        qred[(wv * 2 + h) * 128 + c] = q;
#pragma unroll
        for (int v = 0; v < 8; ++v)
            out[((size_t)b * 128 + wv * 16 + v + 8 * h) * 128 + c] = acc[ct][v];
    }
    __syncthreads();
    if (t < 128) {
        float S = 0.f, Q = 0.f;
        for (int sl = 0; sl < 16; ++sl) { S += sred[sl * 128 + t]; Q += qred[sl * 128 + t]; }
        partS1[(size_t)b * 128 + t] = S;
        partQ1[(size_t)b * 128 + t] = Q;
    }
}

// ---------------------------------------------------------------------------
// K6: in-place BN1 + LeakyReLU epilogue on d_out [B][N][128]
// ---------------------------------------------------------------------------
__global__ __launch_bounds__(256) void k6_epilogue(float* __restrict__ out,
                                                   const float* __restrict__ scale1,
                                                   const float* __restrict__ bias1) {
    __shared__ float s1[128], b1[128];
    const int t = threadIdx.x;
    if (t < 128) { s1[t] = scale1[t]; b1[t] = bias1[t]; }
    __syncthreads();
    size_t vi = (size_t)blockIdx.x * 256 + t;     // float4 index
    float4* p = (float4*)out;
    float4 v = p[vi];
    int c = (int)((vi * 4) & 127);
    float z0 = fmaf(v.x, s1[c + 0], b1[c + 0]);
    float z1 = fmaf(v.y, s1[c + 1], b1[c + 1]);
    float z2 = fmaf(v.z, s1[c + 2], b1[c + 2]);
    float z3 = fmaf(v.w, s1[c + 3], b1[c + 3]);
    v.x = z0 > 0.f ? z0 : 0.01f * z0;
    v.y = z1 > 0.f ? z1 : 0.01f * z1;
    v.z = z2 > 0.f ? z2 : 0.01f * z2;
    v.w = z3 > 0.f ? z3 : 0.01f * z3;
    p[vi] = v;
}

// ---------------------------------------------------------------------------
// Launch
// ---------------------------------------------------------------------------
extern "C" void kernel_launch(void* const* d_in, const int* in_sizes, int n_in,
                              void* d_out, int out_size, void* d_ws, size_t ws_size,
                              hipStream_t stream) {
    (void)in_sizes; (void)n_in; (void)out_size; (void)ws_size;
    const float* node   = (const float*)d_in[0];
    const float* edge   = (const float*)d_in[1];
    const float* w0     = (const float*)d_in[2];
    const float* gamma0 = (const float*)d_in[3];
    const float* beta0  = (const float*)d_in[4];
    const float* w1     = (const float*)d_in[5];
    const float* gamma1 = (const float*)d_in[6];
    const float* beta1  = (const float*)d_in[7];
    float* out = (float*)d_out;
    char*  ws  = (char*)d_ws;

    // Workspace layout (bytes)
    unsigned short* nodeT = (unsigned short*)(ws);                       //  32 MB
    unsigned short* xT    = (unsigned short*)(ws + 33554432ull);         //  96 MB
    unsigned short* y0T   = (unsigned short*)(ws + 134217728ull);        //  64 MB
    unsigned short* w0bf  = (unsigned short*)(ws + 201326592ull);        //  48 KB
    unsigned short* w1bf  = (unsigned short*)(ws + 201375744ull);        //  64 KB
    float* partS0 = (float*)(ws + 201441280ull);                         //   1 MB
    float* partQ0 = (float*)(ws + 202489856ull);                         //   1 MB
    float* partS1 = (float*)(ws + 203538432ull);                         // 512 KB
    float* partQ1 = (float*)(ws + 204062720ull);                         // 512 KB
    float* scale0 = (float*)(ws + 204587008ull);
    float* bias0  = scale0 + 256;
    float* scale1 = bias0 + 256;
    float* bias1  = scale1 + 128;

    const float invN = 1.0f / (1024.0f * 128.0f);

    k_weights<<<64, 256, 0, stream>>>(w0, w1, w0bf, w1bf);
    k0_prep<<<1024, 256, 0, stream>>>(node, xT, nodeT);
    k1_aggr<<<2048, 256, 0, stream>>>(edge, nodeT, xT);
    k2_layer0<<<1024, 256, 0, stream>>>(xT, w0bf, y0T, partS0, partQ0);
    k_finalize<<<1, 256, 0, stream>>>(partS0, partQ0, gamma0, beta0, scale0, bias0,
                                      256, 1024, invN);
    k4_layer1<<<1024, 256, 0, stream>>>(y0T, w1bf, scale0, bias0, out, partS1, partQ1);
    k_finalize<<<1, 128, 0, stream>>>(partS1, partQ1, gamma1, beta1, scale1, bias1,
                                      128, 1024, invN);
    k6_epilogue<<<16384, 256, 0, stream>>>(out, scale1, bias1);
}
